// ATSBlock_58780922413331
// MI455X (gfx1250) — matrix operand order repaired and checked by reading, not compiled, not run on hardware
//
#include <hip/hip_runtime.h>
#include <cmath>

// ---------------- problem constants ----------------
#define HWV   65536
#define BATCH 2
#define HEADS 8
#define HID   170
#define PI2   340
#define GEMM_KPMAX 192   // max padded K (K=170 -> 192, multiple of 32)
#define GEMM_CINMAX 176  // f32 staging rows (>=170)

typedef __attribute__((ext_vector_type(16))) _Float16 v16h;
typedef __attribute__((ext_vector_type(8)))  _Float16 v8h;
typedef __attribute__((ext_vector_type(8)))  float    v8f;
typedef __attribute__((ext_vector_type(4)))  unsigned int u32x4;
typedef __attribute__((ext_vector_type(8)))  unsigned int u32x8;

// Build a 16-half WMMA fragment from two 16B LDS chunks (each -> ds_load_b128).
__device__ __forceinline__ v16h make_frag(const _Float16* lo, const _Float16* hi) {
  v8h a = *(const v8h*)lo;
  v8h b = *(const v8h*)hi;
  v16h r;
#pragma unroll
  for (int i = 0; i < 8; ++i) { r[i] = a[i]; r[i + 8] = b[i]; }
  return r;
}

__device__ __forceinline__ int shifted_pix(int p, int sh, int sw) {
  int h = (p >> 8) + sh;
  int w = (p & 255) + sw;
  return ((h & 255) << 8) | (w & 255);
}

// LDS byte offset of a __shared__ object (for the TDM descriptor).
__device__ __forceinline__ unsigned lds_byte_off(const void* p) {
  return (unsigned)(unsigned long long)
      (__attribute__((address_space(3))) const void*)p;
}

// ---------------------------------------------------------------------------
// Fused conv1x1 GEMM:  out[b,n,p] = sum_k A'[b,k,p] * W[n,k] (+bias) (+res)
// A' = optional LayerNorm(channel) of A, read at torus-shifted pixel (sh,sw).
// A tile (64 contiguous pixels x Cin rows, row stride HW) is DMA'd into LDS by
// the Tensor Data Mover when unshifted; TDM overlaps with f16 weight staging.
// Block: 256 thr (8 waves). Tile: M=64 pixels x N=128 outputs. f16 WMMA, f32 acc.
// ---------------------------------------------------------------------------
__global__ __launch_bounds__(256) void k_gemm(
    const float* __restrict__ in, const float* __restrict__ wgt,
    const float* __restrict__ bias, const float* __restrict__ lnw,
    const float* __restrict__ lnb, const float* __restrict__ res,
    float* __restrict__ out, int Cin, int Kpad, int Cout, int sh, int sw)
{
  __shared__ __align__(16) float Ag[GEMM_CINMAX * 64];   // f32 stage [k][m]
  __shared__ _Float16 Asm[64 * GEMM_KPMAX];              // f16 A   [m][Kpad]
  __shared__ _Float16 Wsm[128 * GEMM_KPMAX];             // f16 W   [n][Kpad]
  __shared__ float    mu_s[64], ri_s[64];
  __shared__ float    Osm[8][16][17];

  const int t  = threadIdx.x;
  const int mb = blockIdx.x;          // B * 1024 m-blocks
  const int b  = mb >> 10;
  const int p0 = (mb & 1023) << 6;    // 64 pixels per block
  const int n0 = blockIdx.y << 7;     // 128 outputs per block
  const int wid = t >> 5, lane = t & 31;

  const bool useTdm = (sh == 0) && (sw == 0);
  if (useTdm && wid == 0) {
    // ---- Tensor DMA: 2D tile, tile_dim0=64 px (contiguous f32),
    //      tile_dim1=Cin rows, row stride HW elements -> LDS contiguous [k][m].
    unsigned long long ga =
        (unsigned long long)(const void*)(in + (size_t)b * Cin * HWV + p0);
    u32x4 g0;
    g0[0] = 1u;                                   // count=1 (valid), user mode
    g0[1] = lds_byte_off(Ag);                     // lds_addr
    g0[2] = (unsigned)ga;                         // global_addr[31:0]
    g0[3] = (unsigned)((ga >> 32) & 0x01FFFFFFu)  // global_addr[56:32]
            | 0x80000000u;                        // type=2 ("image")
    u32x8 g1;
    g1[0] = 0x00020000u;                          // data_size=4B, no multicast
    g1[1] = 0xFFFF0000u;                          // tensor_dim0 lo16 (huge)
    g1[2] = 0xFFFFFFFFu;                          // tensor_dim0 hi / dim1 lo
    g1[3] = 0x0000FFFFu | (64u << 16);            // dim1 hi, tile_dim0=64
    g1[4] = (unsigned)Cin;                        // tile_dim1=Cin, tile_dim2=0
    g1[5] = (unsigned)HWV;                        // tensor_dim0_stride lo32
    g1[6] = 0u;                                   // stride hi / dim1_stride lo
    g1[7] = 0u;
    asm volatile("tensor_load_to_lds %0, %1" :: "s"(g0), "s"(g1) : "memory");
  }

  // ---- stage weights (zero-padded) as f16 [n_local][Kpad]  (overlaps TDM)
  for (int idx = t; idx < 128 * Kpad; idx += 256) {
    int nl = idx / Kpad, k = idx - nl * Kpad;
    int n  = n0 + nl;
    float v = (n < Cout && k < Cin) ? wgt[n * Cin + k] : 0.f;
    Wsm[nl * Kpad + k] = (_Float16)v;
  }

  if (!useTdm) {
    // shifted read wraps the torus -> per-element addressing fallback
    for (int idx = t; idx < 64 * Cin; idx += 256) {
      int c = idx >> 6, m = idx & 63;
      int p = shifted_pix(p0 + m, sh, sw);
      Ag[c * 64 + m] = in[(size_t)(b * Cin + c) * HWV + p];
    }
  }
  if (useTdm && wid == 0) __builtin_amdgcn_s_wait_tensorcnt(0);
  __syncthreads();

  // ---- optional fused LayerNorm stats (Cin==64 when enabled)
  if (lnw) {
    if (t < 64) {
      float s = 0.f, s2 = 0.f;
      for (int c = 0; c < 64; ++c) { float v = Ag[c * 64 + t]; s += v; s2 += v * v; }
      float mu  = s * (1.f / 64.f);
      float var = s2 * (1.f / 64.f) - mu * mu;
      mu_s[t] = mu;
      ri_s[t] = rsqrtf(var + 1e-5f);
    }
    __syncthreads();
  }

  // ---- convert to f16 A fragments layout [m][Kpad]
  for (int idx = t; idx < 64 * Kpad; idx += 256) {
    int m = idx & 63, k = idx >> 6;
    float v = 0.f;
    if (k < Cin) {
      v = Ag[k * 64 + m];
      if (lnw) v = (v - mu_s[m]) * ri_s[m] * lnw[k] + lnb[k];
    }
    Asm[m * Kpad + k] = (_Float16)v;
  }
  __syncthreads();

  // ---- WMMA: each wave owns one 16-wide N tile, loops 4 M tiles
  const int hi = lane >> 4, l16 = lane & 15;
  const int ntile = n0 + wid * 16;
  if (ntile < Cout) {
    for (int mt = 0; mt < 4; ++mt) {
      v8f acc = {};
      for (int kb = 0; kb < Kpad; kb += 32) {
        v16h af = make_frag(&Asm[(mt * 16 + l16) * Kpad + kb + hi * 8],
                            &Asm[(mt * 16 + l16) * Kpad + kb + 16 + hi * 8]);
        v16h bf = make_frag(&Wsm[(wid * 16 + l16) * Kpad + kb + hi * 8],
                            &Wsm[(wid * 16 + l16) * Kpad + kb + 16 + hi * 8]);
        acc = __builtin_amdgcn_wmma_f32_16x16x32_f16(false, af, false, bf,
                                                     (short)0, acc, false, false);
      }
      // transpose through LDS so global stores are pixel-contiguous
#pragma unroll
      for (int r = 0; r < 8; ++r) Osm[wid][l16][r + hi * 8] = acc[r];
      for (int e = lane; e < 256; e += 32) {
        int nl = e >> 4, ml = e & 15;
        int n = ntile + nl;
        if (n < Cout) {
          int p = p0 + mt * 16 + ml;
          size_t o = (size_t)(b * Cout + n) * HWV + p;
          float v = Osm[wid][nl][ml];
          if (bias) v += bias[n];
          if (res)  v += res[o];
          out[o] = v;
        }
      }
    }
  }
}

// ---------------- depthwise 3x3 (SAME, no bias) ----------------
__global__ __launch_bounds__(256) void k_dw(
    const float* __restrict__ in, const float* __restrict__ w9,
    float* __restrict__ out, int C)
{
  int idx = blockIdx.x * 256 + threadIdx.x;
  int p  = idx & (HWV - 1);
  int bc = idx >> 16;
  int c  = bc % C;
  int h = p >> 8, w = p & 255;
  const float* base = in + (size_t)bc * HWV;
  const float* wc = w9 + c * 9;
  __builtin_prefetch(base + ((h + 2) & 255) * 256 + w, 0, 1);  // next row
  float acc = 0.f;
#pragma unroll
  for (int dy = -1; dy <= 1; ++dy) {
    int hh = h + dy; if ((unsigned)hh >= 256u) continue;
#pragma unroll
    for (int dx = -1; dx <= 1; ++dx) {
      int ww = w + dx; if ((unsigned)ww >= 256u) continue;
      acc += wc[(dy + 1) * 3 + (dx + 1)] * base[hh * 256 + ww];
    }
  }
  out[idx] = acc;
}

// ---------------- GDFN fused: depthwise3x3+bias on 340ch, split, gelu*gate ----------------
__global__ __launch_bounds__(256) void k_dwgate(
    const float* __restrict__ in, const float* __restrict__ w9,
    const float* __restrict__ bias, float* __restrict__ out)
{
  int idx = blockIdx.x * 256 + threadIdx.x;   // over B*170*HW
  int p  = idx & (HWV - 1);
  int bc = idx >> 16;
  int b = bc / HID, c = bc - b * HID;
  int h = p >> 8, w = p & 255;
  float a0 = bias[c], a1 = bias[c + HID];
  const float* b0 = in + ((size_t)b * PI2 + c) * HWV;
  const float* b1 = b0 + (size_t)HID * HWV;
  const float* w0 = w9 + c * 9;
  const float* w1 = w9 + (c + HID) * 9;
  __builtin_prefetch(b0 + ((h + 2) & 255) * 256 + w, 0, 1);
  __builtin_prefetch(b1 + ((h + 2) & 255) * 256 + w, 0, 1);
#pragma unroll
  for (int dy = -1; dy <= 1; ++dy) {
    int hh = h + dy; if ((unsigned)hh >= 256u) continue;
#pragma unroll
    for (int dx = -1; dx <= 1; ++dx) {
      int ww = w + dx; if ((unsigned)ww >= 256u) continue;
      int ti = (dy + 1) * 3 + (dx + 1);
      int po = hh * 256 + ww;
      a0 += w0[ti] * b0[po];
      a1 += w1[ti] * b1[po];
    }
  }
  float g = 0.5f * a0 * (1.f + erff(a0 * 0.70710678118f));  // exact gelu
  out[idx] = g * a1;
}

// ---------------- spatial window attention (spa_atsa core) ----------------
// One block per 8x8 window; 8 waves = 8 heads. WMMA 16x16x32 (K=8 zero-padded).
__global__ __launch_bounds__(256) void k_spa_attn(
    const float* __restrict__ qkv,   // [B,192,HW]
    const float* __restrict__ thrw,  // [8,64,9]
    const float* __restrict__ pos,   // [8,64,64]
    const float* __restrict__ wsc,   // [1]
    float* __restrict__ O)           // [B,64,HW]
{
  __shared__ _Float16 qh[8][64][32];   // A layout [tok][Kpad], raw q
  __shared__ _Float16 kh[8][64][32];   // B layout [tok][Kpad], pre-scaled by d^-0.5
  __shared__ _Float16 vT[8][16][64];   // B layout [d(pad16)][tok]
  __shared__ _Float16 at[8][16][64];   // per-wave attn tile, A layout
  __shared__ float thr_s[8][64];

  const int t = threadIdx.x;
  const int blk = blockIdx.x;
  const int b = blk >> 10;
  const int wy = (blk >> 5) & 31, wx = blk & 31;
  const int pbase = ((wy * 8) << 8) + wx * 8;

  for (int idx = t; idx < 8 * 64 * 32; idx += 256)
    if ((idx & 31) >= 8) { ((_Float16*)qh)[idx] = (_Float16)0.f; ((_Float16*)kh)[idx] = (_Float16)0.f; }
  for (int idx = t; idx < 8 * 16 * 64; idx += 256)
    if (((idx >> 6) & 15) >= 8) ((_Float16*)vT)[idx] = (_Float16)0.f;

  const float kscale = 0.35355339059327373f;  // 8^-0.5 folded into K
  for (int idx = t; idx < 4096; idx += 256) {
    int tok = idx & 63, c = idx >> 6;
    int hd = c >> 3, d = c & 7;
    int p = pbase + ((tok >> 3) << 8) + (tok & 7);
    size_t base = (size_t)(b * 192 + c) * HWV + p;
    qh[hd][tok][d] = (_Float16)qkv[base];
    kh[hd][tok][d] = (_Float16)(qkv[base + (size_t)64 * HWV] * kscale);
    vT[hd][d][tok] = (_Float16)qkv[base + (size_t)128 * HWV];
  }
  __syncthreads();

  // threshold = lrelu(conv3x3(q_raw, thr_w)) per head per token (zero pad in-window)
  for (int o = t; o < 512; o += 256) {
    int hd = o >> 6, tok = o & 63;
    int ty = tok >> 3, tx = tok & 7;
    float acc = 0.f;
    for (int c = 0; c < 64; ++c) {
      const float* wr = thrw + (hd * 64 + c) * 9;
      const _Float16* qc = &qh[c >> 3][0][c & 7];
#pragma unroll
      for (int dy = -1; dy <= 1; ++dy) {
        int yy = ty + dy; if ((unsigned)yy >= 8u) continue;
#pragma unroll
        for (int dx = -1; dx <= 1; ++dx) {
          int xx = tx + dx; if ((unsigned)xx >= 8u) continue;
          acc += wr[(dy + 1) * 3 + (dx + 1)] * (float)qc[(yy * 8 + xx) * 32];
        }
      }
    }
    thr_s[hd][tok] = acc >= 0.f ? acc : 0.01f * acc;
  }
  __syncthreads();

  const float wm = wsc[0], wm2 = 1.f - wm;
  const int wid = t >> 5, lane = t & 31;
  const int hi = lane >> 4, l16 = lane & 15;
  const int hd = wid;

  for (int mt = 0; mt < 4; ++mt) {
    v8f st[4];
#pragma unroll
    for (int nt = 0; nt < 4; ++nt) {
      v8f acc = {};
      v16h af = make_frag(&qh[hd][mt * 16 + l16][hi * 8],
                          &qh[hd][mt * 16 + l16][16 + hi * 8]);
      v16h bf = make_frag(&kh[hd][nt * 16 + l16][hi * 8],
                          &kh[hd][nt * 16 + l16][16 + hi * 8]);
      acc = __builtin_amdgcn_wmma_f32_16x16x32_f16(false, af, false, bf,
                                                   (short)0, acc, false, false);
      st[nt] = acc;
    }
    float sv[4][8], tv[4][8];
#pragma unroll
    for (int nt = 0; nt < 4; ++nt)
#pragma unroll
      for (int r = 0; r < 8; ++r) {
        int m = mt * 16 + r + hi * 8;
        int n = nt * 16 + l16;
        float raw = st[nt][r];
        float pv = pos[(hd * 64 + m) * 64 + n];
        sv[nt][r] = raw + pv;
        tv[nt][r] = (raw < thr_s[hd][m] ? 0.f : raw) + pv;
      }
    // softmax over 64 columns: rows live on 16-lane half groups (D layout)
#pragma unroll
    for (int r = 0; r < 8; ++r) {
      float mx = fmaxf(fmaxf(sv[0][r], sv[1][r]), fmaxf(sv[2][r], sv[3][r]));
      for (int m = 1; m <= 8; m <<= 1) mx = fmaxf(mx, __shfl_xor(mx, m, 32));
      float ex[4], sm = 0.f;
#pragma unroll
      for (int nt = 0; nt < 4; ++nt) { ex[nt] = __expf(sv[nt][r] - mx); sm += ex[nt]; }
      for (int m = 1; m <= 8; m <<= 1) sm += __shfl_xor(sm, m, 32);
      float rs = wm / sm;
#pragma unroll
      for (int nt = 0; nt < 4; ++nt)
        at[wid][r + hi * 8][nt * 16 + l16] = (_Float16)(ex[nt] * rs + tv[nt][r] * wm2);
    }
    // o = attn @ V : M=16 tok, N=16(d pad), K=64
    v8f oacc = {};
    for (int kb = 0; kb < 64; kb += 32) {
      v16h af = make_frag(&at[wid][l16][kb + hi * 8], &at[wid][l16][kb + 16 + hi * 8]);
      v16h bf = make_frag(&vT[hd][l16][kb + hi * 8], &vT[hd][l16][kb + 16 + hi * 8]);
      oacc = __builtin_amdgcn_wmma_f32_16x16x32_f16(false, af, false, bf,
                                                    (short)0, oacc, false, false);
    }
    if (l16 < 8) {
      int cch = hd * 8 + l16;
#pragma unroll
      for (int r = 0; r < 8; ++r) {
        int tok = mt * 16 + r + hi * 8;
        int p = pbase + ((tok >> 3) << 8) + (tok & 7);
        O[(size_t)(b * 64 + cch) * HWV + p] = oacc[r];
      }
    }
  }
}

// ---------------- spectral (channel) window attention (spe_atsa core) ----------------
// One block per window; 4 waves each own a 16-channel M tile. K=64 (pixels).
// Threshold mean(conv3x3) collapsed to 9 rectangle sums per channel.
__global__ __launch_bounds__(128) void k_spe_attn(
    const float* __restrict__ qkv,   // [B,192,HW] already in rolled coords
    const float* __restrict__ thrw,  // [64,64,9]
    const float* __restrict__ thrb,  // [64]
    const float* __restrict__ pos,   // [64,64]
    const float* __restrict__ wsc,
    float* __restrict__ O)           // [B,64,HW] rolled coords
{
  __shared__ _Float16 qS[64][64];    // A layout [c][pix], raw
  __shared__ _Float16 kS[64][64];    // B layout [c_j][pix], pre-scaled
  __shared__ _Float16 vT[64][64];    // B layout [pix][c_j]
  __shared__ _Float16 at[4][16][64];
  __shared__ float Tc[64], Rr[64][8], Cc[64][8], thr_v[64];

  const int t = threadIdx.x;
  const int blk = blockIdx.x;
  const int b = blk >> 10;
  const int wy = (blk >> 5) & 31, wx = blk & 31;
  const int pbase = ((wy * 8) << 8) + wx * 8;
  const float kscale = 0.35355339059327373f;

  for (int idx = t; idx < 4096; idx += 128) {
    int tok = idx & 63, c = idx >> 6;
    int p = pbase + ((tok >> 3) << 8) + (tok & 7);
    size_t base = (size_t)(b * 192 + c) * HWV + p;
    qS[c][tok] = (_Float16)qkv[base];
    kS[c][tok] = (_Float16)(qkv[base + (size_t)64 * HWV] * kscale);
    vT[tok][c] = (_Float16)qkv[base + (size_t)128 * HWV];
  }
  __syncthreads();

  if (t < 64) {             // per-channel total / row / col sums of q
    float tt = 0.f, rr[8] = {}, cc[8] = {};
    for (int tok = 0; tok < 64; ++tok) {
      float v = (float)qS[t][tok];
      tt += v; rr[tok >> 3] += v; cc[tok & 7] += v;
    }
    Tc[t] = tt;
#pragma unroll
    for (int i = 0; i < 8; ++i) { Rr[t][i] = rr[i]; Cc[t][i] = cc[i]; }
  }
  __syncthreads();

  if (t < 64) {             // thr_o = lrelu(b_o + (1/64) sum_{c,tap} w * rectsum)
    int o = t;
    float s = 0.f;
    for (int c = 0; c < 64; ++c) {
      const float* wr = thrw + (o * 64 + c) * 9;
      float Tv = Tc[c];
#pragma unroll
      for (int dy = -1; dy <= 1; ++dy)
#pragma unroll
        for (int dx = -1; dx <= 1; ++dx) {
          float S = Tv;
          int er = -1, ec = -1;
          if (dy == -1) { S -= Rr[c][7]; er = 7; } else if (dy == 1) { S -= Rr[c][0]; er = 0; }
          if (dx == -1) { S -= Cc[c][7]; ec = 7; } else if (dx == 1) { S -= Cc[c][0]; ec = 0; }
          if (er >= 0 && ec >= 0) S += (float)qS[c][er * 8 + ec];
          s += wr[(dy + 1) * 3 + (dx + 1)] * S;
        }
    }
    float th = thrb[o] + s * (1.f / 64.f);
    thr_v[o] = th >= 0.f ? th : 0.01f * th;
  }
  __syncthreads();

  const float wm = wsc[0], wm2 = 1.f - wm;
  const int wid = t >> 5, lane = t & 31;
  const int hi = lane >> 4, l16 = lane & 15;
  const int mt = wid;

  v8f st[4];
#pragma unroll
  for (int nt = 0; nt < 4; ++nt) {
    v8f acc = {};
    for (int kb = 0; kb < 64; kb += 32) {
      v16h af = make_frag(&qS[mt * 16 + l16][kb + hi * 8], &qS[mt * 16 + l16][kb + 16 + hi * 8]);
      v16h bf = make_frag(&kS[nt * 16 + l16][kb + hi * 8], &kS[nt * 16 + l16][kb + 16 + hi * 8]);
      acc = __builtin_amdgcn_wmma_f32_16x16x32_f16(false, af, false, bf,
                                                   (short)0, acc, false, false);
    }
    st[nt] = acc;
  }
  float sv[4][8], tv[4][8];
#pragma unroll
  for (int nt = 0; nt < 4; ++nt)
#pragma unroll
    for (int r = 0; r < 8; ++r) {
      int m = mt * 16 + r + hi * 8;
      int n = nt * 16 + l16;
      float raw = st[nt][r];
      float pv = pos[m * 64 + n];
      sv[nt][r] = raw + pv;
      tv[nt][r] = (raw < thr_v[m] ? 0.f : raw) + pv;
    }
#pragma unroll
  for (int r = 0; r < 8; ++r) {
    float mx = fmaxf(fmaxf(sv[0][r], sv[1][r]), fmaxf(sv[2][r], sv[3][r]));
    for (int m = 1; m <= 8; m <<= 1) mx = fmaxf(mx, __shfl_xor(mx, m, 32));
    float ex[4], sm = 0.f;
#pragma unroll
    for (int nt = 0; nt < 4; ++nt) { ex[nt] = __expf(sv[nt][r] - mx); sm += ex[nt]; }
    for (int m = 1; m <= 8; m <<= 1) sm += __shfl_xor(sm, m, 32);
    float rs = wm / sm;
#pragma unroll
    for (int nt = 0; nt < 4; ++nt)
      at[wid][r + hi * 8][nt * 16 + l16] = (_Float16)(ex[nt] * rs + tv[nt][r] * wm2);
  }
  // o = attn @ V : M=16 ch, N=64 pix, K=64 ch
  for (int nt = 0; nt < 4; ++nt) {
    v8f oacc = {};
    for (int kb = 0; kb < 64; kb += 32) {
      v16h af = make_frag(&at[wid][l16][kb + hi * 8], &at[wid][l16][kb + 16 + hi * 8]);
      v16h bf = make_frag(&vT[nt * 16 + l16][kb + hi * 8], &vT[nt * 16 + l16][kb + 16 + hi * 8]);
      oacc = __builtin_amdgcn_wmma_f32_16x16x32_f16(false, af, false, bf,
                                                    (short)0, oacc, false, false);
    }
#pragma unroll
    for (int r = 0; r < 8; ++r) {
      int cch = mt * 16 + r + hi * 8;
      int tok = nt * 16 + l16;
      int p = pbase + ((tok >> 3) << 8) + (tok & 7);
      O[(size_t)(b * 64 + cch) * HWV + p] = oacc[r];
    }
  }
}

// ---------------------------------------------------------------------------
extern "C" void kernel_launch(void* const* d_in, const int* in_sizes, int n_in,
                              void* d_out, int out_size, void* d_ws, size_t ws_size,
                              hipStream_t stream)
{
  (void)in_sizes; (void)n_in; (void)out_size; (void)ws_size;
  const float* const* in = (const float* const*)d_in;
  float* xout = (float*)d_out;
  char* ws = (char*)d_ws;
  size_t off = 0;
  auto alloc = [&](size_t bytes) {
    char* p = ws + off; off += (bytes + 255) & ~(size_t)255; return (float*)p;
  };
  float* buf1 = alloc((size_t)BATCH * PI2 * HWV * 4);   // up to 340 ch
  float* buf2 = alloc((size_t)BATCH * 192 * HWV * 4);   // up to 192 ch
  float* buf3 = alloc((size_t)BATCH * 64  * HWV * 4);   // attention output

  auto gemm = [&](const float* A, const float* W, const float* bias,
                  const float* lw, const float* lb, const float* res,
                  float* out_, int Cin, int Kpad, int Cout, int sh, int sw) {
    dim3 g(BATCH * 1024, (Cout + 127) / 128);
    k_gemm<<<g, 256, 0, stream>>>(A, W, bias, lw, lb, res, out_, Cin, Kpad, Cout, sh, sw);
  };

  const float* xin = in[0];   // input image; d_out becomes the running residual
  for (int i = 0; i < 2; ++i) {
    const int P = 1 + i * 31;  // params in setup_inputs() insertion order
    const float *ln0w = in[P+0], *ln0b = in[P+1], *ln1w = in[P+2], *ln1b = in[P+3];
    const float *ln2w = in[P+4], *ln2b = in[P+5], *ln3w = in[P+6], *ln3b = in[P+7];
    const float *spa_qkv = in[P+8],  *spa_dw = in[P+9],  *spa_thr = in[P+10];
    const float *spa_pos = in[P+11], *spa_w  = in[P+12], *spa_pow = in[P+13], *spa_pob = in[P+14];
    const float *spe_qkv = in[P+15], *spe_dw = in[P+16], *spe_thr = in[P+17], *spe_thrb = in[P+18];
    const float *spe_pos = in[P+19], *spe_w  = in[P+20], *spe_pow = in[P+21], *spe_pob = in[P+22];
    const float *f0_pi = in[P+23], *f0_dw = in[P+24], *f0_db = in[P+25], *f0_po = in[P+26];
    const float *f1_pi = in[P+27], *f1_dw = in[P+28], *f1_db = in[P+29], *f1_po = in[P+30];

    // ---- spa: LN0 + qkv, depthwise, window attention, po + residual
    gemm(xin, spa_qkv, nullptr, ln0w, ln0b, nullptr, buf1, 64, 64, 192, 0, 0);
    k_dw<<<BATCH * 192 * HWV / 256, 256, 0, stream>>>(buf1, spa_dw, buf2, 192);
    k_spa_attn<<<BATCH * 1024, 256, 0, stream>>>(buf2, spa_thr, spa_pos, spa_w, buf3);
    gemm(buf3, spa_pow, spa_pob, nullptr, nullptr, xin, xout, 64, 64, 64, 0, 0);
    xin = xout;

    // ---- ffn0: LN1 + pi, dw+gelu-gate fused, po + residual
    gemm(xin, f0_pi, nullptr, ln1w, ln1b, nullptr, buf1, 64, 64, PI2, 0, 0);
    k_dwgate<<<BATCH * HID * HWV / 256, 256, 0, stream>>>(buf1, f0_dw, f0_db, buf2);
    gemm(buf2, f0_po, nullptr, nullptr, nullptr, xin, xout, HID, 192, 64, 0, 0);

    // ---- spe: LN2 + roll(+4) folded into qkv read, dw, channel attention,
    //           po with roll(-4) folded into its read, + residual
    gemm(xin, spe_qkv, nullptr, ln2w, ln2b, nullptr, buf1, 64, 64, 192, -4, -4);
    k_dw<<<BATCH * 192 * HWV / 256, 256, 0, stream>>>(buf1, spe_dw, buf2, 192);
    k_spe_attn<<<BATCH * 1024, 128, 0, stream>>>(buf2, spe_thr, spe_thrb, spe_pos, spe_w, buf3);
    gemm(buf3, spe_pow, spe_pob, nullptr, nullptr, xin, xout, 64, 64, 64, 4, 4);

    // ---- ffn1
    gemm(xin, f1_pi, nullptr, ln3w, ln3b, nullptr, buf1, 64, 64, PI2, 0, 0);
    k_dwgate<<<BATCH * HID * HWV / 256, 256, 0, stream>>>(buf1, f1_dw, f1_db, buf2);
    gemm(buf2, f1_po, nullptr, nullptr, nullptr, xin, xout, HID, 192, 64, 0, 0);
  }
}
